// SeqAttention_42228118454969
// MI455X (gfx1250) — matrix operand order repaired and checked by reading, not compiled, not run on hardware
//
#include <hip/hip_runtime.h>
#include <hip/hip_bf16.h>

// ---- problem constants (from reference) ----
#define NB_HEADS 12
#define HEAD_DIM 64
#define BLOCK_M  512
#define SPAN     2048
#define RAMPF    64.0f
#define SKIP     960          // computed skip length
#define NKV      1600         // (BLOCK+SPAN) - SKIP : sliced K/V length
#define LC       1088         // SPAN - SKIP : relative window length
#define NT16     69           // 16-wide tiles covering band width 1104
#define LT16     68           // 16-wide tiles covering LC=1088
#define SW       1120         // padded score-buffer width (35 * 32)
#define KT32     35           // 32-wide k chunks over SW

// ---- LDS layout (float indices) ----
#define SBUF_F   (16 * SW)                 // 17920
#define QS_F     (16 * HEAD_DIM)           // 1024
#define RED_F    (16 + 16 + 128 + 128)     // 288
#define STAGE_OFF_F (SBUF_F + QS_F + RED_F)// 19232
#define STAGE_F  8192                      // 32KB shared staging (PE or V)
#define TOTAL_F  (STAGE_OFF_F + STAGE_F)   // 27424 floats = 109696 bytes

typedef __attribute__((ext_vector_type(16))) __bf16 v16bf;
typedef __attribute__((ext_vector_type(8)))  float  v8f;
typedef unsigned int v4u __attribute__((ext_vector_type(4)));
typedef int v8i __attribute__((ext_vector_type(8)));
typedef int v4i __attribute__((ext_vector_type(4)));

__device__ __forceinline__ v8f wmma_bf16(v16bf a, v16bf b, v8f c) {
  // (neg_a, A, neg_b, B, c_mod, C, reuse_a, reuse_b)
  return __builtin_amdgcn_wmma_f32_16x16x32_bf16(false, a, false, b, (short)0, c,
                                                 false, false);
}

// TDM 2-D tile load: Memory[gtile + (y*stride0 + x)*4] -> LDS[lds_off + (y*tile_d0 + x)*4]
// Dims/strides in 4-byte elements. Rows >= tensor_d1 read as zero (OOB fill).
// D# packing per CDNA5 ISA.
__device__ __forceinline__ void tdm_load_2d(unsigned lds_byte_off,
                                            const float* gtile,
                                            unsigned tensor_d0, unsigned tensor_d1,
                                            unsigned tile_d0, unsigned tile_d1,
                                            unsigned long long stride0)
{
  const unsigned long long ga = (unsigned long long)(const void*)gtile;
  v4u g0 = {};
  g0[0] = 1u;                                           // count=1, user mode
  g0[1] = lds_byte_off;                                 // lds_addr
  g0[2] = (unsigned)ga;                                 // global_addr[31:0]
  g0[3] = (unsigned)((ga >> 32) & 0x01ffffffu)          // global_addr[56:32]
        | (2u << 30);                                   // type = 2 ("image")

  v8i g1 = {};
  g1[0] = (int)(2u << 16);                              // data_size = 2 -> 4 bytes
  g1[1] = (int)((tensor_d0 & 0xffffu) << 16);           // tensor_dim0[15:0]
  g1[2] = (int)(((tensor_d0 >> 16) & 0xffffu) | ((tensor_d1 & 0xffffu) << 16));
  g1[3] = (int)(((tensor_d1 >> 16) & 0xffffu) | ((tile_d0 & 0xffffu) << 16));
  g1[4] = (int)(tile_d1 & 0xffffu);                     // tile_dim1; tile_dim2=0
  g1[5] = (int)(unsigned)(stride0 & 0xffffffffu);       // tensor_dim0_stride[31:0]
  g1[6] = (int)(unsigned)((stride0 >> 32) & 0xffffu);   // stride[47:32]; dim1_stride=0
  g1[7] = 0;

  v4i g2 = {};
  v4i g3 = {};
#if defined(__clang_major__) && (__clang_major__ >= 23)
  v8i g4 = {};
  __builtin_amdgcn_tensor_load_to_lds(g0, g1, g2, g3, g4, 0);
#else
  __builtin_amdgcn_tensor_load_to_lds(g0, g1, g2, g3, 0);
#endif
}

__global__ void __launch_bounds__(128)
seq_attn_kernel(const float* __restrict__ q,
                const float* __restrict__ kmat,
                const float* __restrict__ vmat,
                const float* __restrict__ pe,
                const float* __restrict__ ratio,
                float* __restrict__ out)
{
  extern __shared__ float smem[];
  float* Sbuf = smem;                    // 16 * SW      (scores -> weights)
  float* Qs   = Sbuf + SBUF_F;           // 16 * 64      (query tile, f32)
  float* rmax = Qs + QS_F;               // 16
  float* invd = rmax + 16;               // 16
  float* redA = invd + 16;               // 128
  float* redB = redA + 128;              // 128
  float* stage = smem + STAGE_OFF_F;     // 8192 floats: PE (phase C) / V (phase E)

  const int mt   = blockIdx.x;
  const int b    = blockIdx.y;
  const int m0   = mt * 16;
  const int h    = b % NB_HEADS;
  const int tid  = threadIdx.x;
  const int lane = tid & 31;
  const int wv   = tid >> 5;             // wave id 0..3 (wave32)
  const int nl   = lane & 15;            // B/C/D column within a 16-wide tile
  const int hi   = lane >> 4;            // half-wave select
  const int akb  = hi << 3;              // A k-offset (0 or 8)
  const int bkb  = hi << 4;              // B k-offset (0 or 16)
  const int mrow = nl;                   // A row = lane & 15

  // dynamic LDS starts after static LDS; stage region byte base
  const unsigned stage_base =
      __builtin_amdgcn_groupstaticsize() + (unsigned)STAGE_OFF_F * 4u;

  // ---------- load Q tile (16 x 64, contiguous) into LDS ----------
  const float* qbase = q + ((size_t)b * BLOCK_M + m0) * HEAD_DIM;
  for (int i = tid; i < 16 * HEAD_DIM; i += 128) Qs[i] = qbase[i];
  __syncthreads();

  // A fragments for Q (k chunks 0..31 and 32..63), shared by QK and Q*PE
  v16bf aq0, aq1;
#pragma unroll
  for (int e = 0; e < 16; ++e) {
    const int kk = bkb + e;                           // contiguous k = 16*half + e (matches B)
    aq0[e] = (__bf16)Qs[mrow * HEAD_DIM + kk];
    aq1[e] = (__bf16)Qs[mrow * HEAD_DIM + 32 + kk];
  }

  // ---------- Phase B: S(abs) = Q * K^T over the diagonal band ----------
  const float* kb_ = kmat + ((size_t)b * (BLOCK_M + SPAN) + SKIP) * HEAD_DIM;
  for (int jt = wv; jt < NT16; jt += 4) {
    const int n0 = m0 + jt * 16;                        // n0+15 <= 1599, in range
    const float* krow = kb_ + (size_t)(n0 + nl) * HEAD_DIM;
    v16bf b0, b1;
#pragma unroll
    for (int e = 0; e < 16; ++e) {                      // B[k,n] = K[n0+n][k]
      b0[e] = (__bf16)krow[bkb + e];
      b1[e] = (__bf16)krow[32 + bkb + e];
    }
    v8f c = {};
    c = wmma_bf16(aq0, b0, c);
    c = wmma_bf16(aq1, b1, c);
#pragma unroll
    for (int j = 0; j < 8; ++j)
      Sbuf[(j + (hi << 3)) * SW + jt * 16 + nl] = c[j];
  }
  __syncthreads();

  // ---------- Phase C: pos = Q * PE (relative), PE tiles via TDM, scatter-add ----------
  {
    float* pew = stage + wv * 2048;                     // two 1024-float buffers/wave
    const unsigned pew_off = stage_base + (unsigned)wv * 8192u;
    int pbuf = 0;
    const int lt0 = wv;
    if (lt0 < LT16)                                     // 64x16 tile, row stride 2048
      tdm_load_2d(pew_off, pe + SKIP + lt0 * 16, 16, 64, 16, 64,
                  (unsigned long long)SPAN);
    for (int lt = lt0; lt < LT16; lt += 4, pbuf ^= 1) {
      const int ltn = lt + 4;
      if (ltn < LT16) {
        tdm_load_2d(pew_off + (unsigned)(pbuf ^ 1) * 4096u,
                    pe + SKIP + ltn * 16, 16, 64, 16, 64,
                    (unsigned long long)SPAN);
        __builtin_amdgcn_s_wait_tensorcnt((short)1);    // older tile done (in-order)
      } else {
        __builtin_amdgcn_s_wait_tensorcnt((short)0);
      }
      const float* pes = pew + pbuf * 1024;             // [k][l] = 64 x 16
      v16bf pb0, pb1;
#pragma unroll
      for (int e = 0; e < 16; ++e) {
        pb0[e] = (__bf16)pes[(bkb + e) * 16 + nl];
        pb1[e] = (__bf16)pes[(32 + bkb + e) * 16 + nl];
      }
      v8f c = {};
      c = wmma_bf16(aq0, pb0, c);
      c = wmma_bf16(aq1, pb1, c);
      const int l = lt * 16 + nl;
#pragma unroll
      for (int j = 0; j < 8; ++j) {
        const int mi = j + (hi << 3);
        Sbuf[mi * SW + l + mi] += c[j];                 // skew: j = l + mi
      }
    }
  }
  __syncthreads();

  // ---------- Phase D: softmax + ramp mask + renorm (exact renorm identity) ----------
  const float shift = ratio[h] * (float)SPAN;
  const int r  = tid & 15;                              // row
  const int c8 = tid >> 4;                              // 0..7 column group
  float pmax = -3.0e38f;
  for (int j = c8; j < SW; j += 8) {
    const int l = j - r;
    if ((unsigned)l < (unsigned)LC)
      pmax = fmaxf(pmax, Sbuf[r * SW + j] * 0.125f);
  }
  redA[r * 8 + c8] = pmax;
  __syncthreads();
  if (tid < 16) {
    float m = redA[tid * 8];
#pragma unroll
    for (int c = 1; c < 8; ++c) m = fmaxf(m, redA[tid * 8 + c]);
    rmax[tid] = m;
  }
  __syncthreads();

  float se = 0.f, sem = 0.f;
  const float rm = rmax[r];
  for (int j = c8; j < SW; j += 8) {
    const int l = j - r;
    float w = 0.f;
    if ((unsigned)l < (unsigned)LC) {
      const float e = __expf(Sbuf[r * SW + j] * 0.125f - rm);
      float msk = ((float)(l - (SPAN - 1 - SKIP)) + shift) / RAMPF + 1.0f;
      msk = fminf(fmaxf(msk, 0.f), 1.f);
      se += e;
      w = e * msk;
      sem += w;
    }
    Sbuf[r * SW + j] = w;                               // overwrite with e*mask
  }
  redA[r * 8 + c8] = se;
  redB[r * 8 + c8] = sem;
  __syncthreads();
  if (tid < 16) {
    float tse = 0.f, tsem = 0.f;
#pragma unroll
    for (int c = 0; c < 8; ++c) { tse += redA[tid * 8 + c]; tsem += redB[tid * 8 + c]; }
    // softmax attn = e/Z ; final = em / (sum_em + 1e-8*Z), Z = sum_e
    invd[tid] = 1.0f / (tsem + 1e-8f * tse);
  }
  __syncthreads();

  // ---------- Phase E: O = P * V ; shared V tiles via TDM, double-buffered ----------
  const float* vb_ = vmat + ((size_t)b * (BLOCK_M + SPAN) + SKIP) * HEAD_DIM;
  const int d0 = wv * 16;                               // each wave owns one d-tile
  const float idn = invd[mrow];
  if (wv == 0)                                          // prologue: first 32x64 V tile
    tdm_load_2d(stage_base, vb_ + (size_t)m0 * HEAD_DIM,
                64, (unsigned)(NKV - m0), 64, 32, (unsigned long long)HEAD_DIM);
  v8f acc = {};
  for (int kt = 0; kt < KT32; ++kt) {
    if (wv == 0) {
      if (kt + 1 < KT32) {
        const int n1 = m0 + (kt + 1) * 32;              // OOB rows zero-filled by TDM
        tdm_load_2d(stage_base + (unsigned)((kt + 1) & 1) * 8192u,
                    vb_ + (size_t)n1 * HEAD_DIM,
                    64, (unsigned)(NKV - n1), 64, 32, (unsigned long long)HEAD_DIM);
        __builtin_amdgcn_s_wait_tensorcnt((short)1);    // tile kt done (in-order)
      } else {
        __builtin_amdgcn_s_wait_tensorcnt((short)0);
      }
    }
    __syncthreads();                                    // publish V tile to all waves
    const float* vs = stage + (kt & 1) * 2048;          // [n][d] = 32 x 64
    const int jb = kt * 32;
    v16bf aP, bV;
#pragma unroll
    for (int e = 0; e < 16; ++e) {
      const int kk = bkb + e;
      aP[e] = (__bf16)(Sbuf[mrow * SW + jb + kk] * idn);
    }
#pragma unroll
    for (int e = 0; e < 16; ++e)
      bV[e] = (__bf16)vs[(bkb + e) * HEAD_DIM + d0 + nl];
    acc = wmma_bf16(aP, bV, acc);
    __syncthreads();                                    // all reads done before overwrite
  }

  float* ob = out + ((size_t)b * BLOCK_M + m0) * HEAD_DIM + d0;
#pragma unroll
  for (int j = 0; j < 8; ++j)
    ob[(size_t)(j + (hi << 3)) * HEAD_DIM + nl] = acc[j];
}

extern "C" void kernel_launch(void* const* d_in, const int* in_sizes, int n_in,
                              void* d_out, int out_size, void* d_ws, size_t ws_size,
                              hipStream_t stream) {
  const float* q     = (const float*)d_in[0];
  const float* k     = (const float*)d_in[1];
  const float* v     = (const float*)d_in[2];
  const float* pe    = (const float*)d_in[3];
  const float* ratio = (const float*)d_in[4];
  float* out = (float*)d_out;

  const int Btot = in_sizes[0] / (BLOCK_M * HEAD_DIM);  // 96 batch-heads
  const size_t smem_bytes = (size_t)TOTAL_F * sizeof(float);

  hipFuncSetAttribute((const void*)seq_attn_kernel,
                      hipFuncAttributeMaxDynamicSharedMemorySize,
                      (int)smem_bytes);

  dim3 grid(BLOCK_M / 16, Btot);
  hipLaunchKernelGGL(seq_attn_kernel, grid, dim3(128), smem_bytes, stream,
                     q, k, v, pe, ratio, out);
}
